// CrossAttentionReader_61194694033984
// MI455X (gfx1250) — matrix operand order repaired and checked by reading, not compiled, not run on hardware
//
#include <hip/hip_runtime.h>
#include <hip/hip_bf16.h>
#include <math.h>

// Problem constants (reference: B=4,N=512 -> BN=2048; M=32768; D=1024; H=16; hd=64; K=32)
#define D_DIM   1024
#define BN      2048
#define M_MEM   32768
#define H_HEADS 16
#define HD      64
#define TOPK_K  32

#define BT_M 64     // block tile rows (big GEMM)
#define BT_N 128    // block tile cols (big GEMM)

typedef __attribute__((ext_vector_type(16))) __bf16 bf16x16;
typedef __attribute__((ext_vector_type(8)))  float  f32x8;

union Frag16 { bf16x16 v; unsigned int w[8]; };

// f32 -> bf16 (round-to-nearest-even), packed pair into one dword
__device__ __forceinline__ unsigned int pack2bf(float a, float b) {
  unsigned int ua = __float_as_uint(a);
  ua += 0x7FFFu + ((ua >> 16) & 1u);
  unsigned int ub = __float_as_uint(b);
  ub += 0x7FFFu + ((ub >> 16) & 1u);
  return (ua >> 16) | (ub & 0xFFFF0000u);
}

__device__ __forceinline__ f32x8 wmma_bf16(const Frag16& a, const Frag16& b, f32x8 c) {
  return __builtin_amdgcn_wmma_f32_16x16x32_bf16(false, a.v, false, b.v,
                                                 (short)0, c, false, false);
}

// CDNA5 async copy global -> LDS, 16 bytes, tracked by ASYNCcnt (no VGPR round-trip).
__device__ __forceinline__ void async_b128(unsigned lds_addr, const void* gaddr) {
  asm volatile("global_load_async_to_lds_b128 %0, %1, off"
               :: "v"(lds_addr), "v"(gaddr) : "memory");
}

// ---------------------------------------------------------------------------
// CDNA5 wave32 fragment layouts (CDNA5 ISA):
//   A 16x32 bf16: lane m=lane&15, half=lane>>4; VGPR j K-pair base = 2j+(j>=4?8:0)+8*half
//   B 32x16 bf16: lane n=lane&15, half=lane>>4; VGPR j K-pair base = 2j+(j>=4?8:0)+8*half (same as A)
//   C/D 16x16 f32: acc[j] = D[j+8*half][n]
// ---------------------------------------------------------------------------

// Big GEMM on pre-converted bf16 operands:
//   C[row,col] = sum_d A[row,d]*W[col,d] (+bias)
// 8 waves/block, 64x128 block tile, each wave computes 32x32 (4 v_wmma / K-step).
// K-tiles staged with GLOBAL_LOAD_ASYNC_TO_LDS_B128 into a double buffer;
// s_wait_asynccnt 3 overlaps the next tile's DMA with the current tile's WMMA.
__global__ __launch_bounds__(256)
void wmma_gemm_bf16_async(const unsigned short* __restrict__ Abf,
                          const unsigned short* __restrict__ Wbf,
                          const float* __restrict__ bias, float* __restrict__ C,
                          int Kdim, int ldC)
{
  __shared__ unsigned int ldsA[2][BT_M * 16];   // 2 x 4 KB, [row][k-pair dword]
  __shared__ unsigned int ldsB[2][BT_N * 16];   // 2 x 8 KB
  const int t    = threadIdx.x;
  const int lane = t & 31;
  const int wave = t >> 5;
  const int waveM = wave & 1;       // 2 row groups of 32
  const int waveN = wave >> 1;      // 4 col groups of 32
  const int rowBase = blockIdx.y * BT_M;
  const int colBase = blockIdx.x * BT_N;

  // staging: A 64 rows x 4x8 bf16 (16B each), B 128 rows x 2x16 bf16 (32B each)
  const int aRowS = t >> 2, aSeg = t & 3;
  const int bRowS = t >> 1, bSeg = t & 1;
  const unsigned short* aSrc = Abf + (size_t)(rowBase + aRowS) * Kdim + aSeg * 8;
  const unsigned short* bSrc = Wbf + (size_t)(colBase + bRowS) * Kdim + bSeg * 16;

  unsigned aDst[2], bDst[2];
#pragma unroll
  for (int b = 0; b < 2; ++b) {
    aDst[b] = (unsigned)(size_t)&ldsA[b][aRowS * 16 + aSeg * 4];
    bDst[b] = (unsigned)(size_t)&ldsB[b][bRowS * 16 + bSeg * 8];
  }

  const int m = lane & 15, half = lane >> 4;
  f32x8 acc00 = {}, acc01 = {}, acc10 = {}, acc11 = {};
  const int nTiles = Kdim / 32;

  // prologue: tile 0 -> buffer 0
  async_b128(aDst[0], aSrc);
  async_b128(bDst[0], bSrc);
  async_b128(bDst[0] + 16, bSrc + 8);

  for (int i = 0; i < nTiles; ++i) {
    const int cur = i & 1;
    if (i + 1 < nTiles) {
      const int nxt = cur ^ 1;
      const unsigned short* aN = aSrc + (size_t)(i + 1) * 32;
      const unsigned short* bN = bSrc + (size_t)(i + 1) * 32;
      async_b128(aDst[nxt], aN);
      async_b128(bDst[nxt], bN);
      async_b128(bDst[nxt] + 16, bN + 8);
      asm volatile("s_wait_asynccnt 0x3" ::: "memory");  // tile i complete (in-order)
    } else {
      asm volatile("s_wait_asynccnt 0x0" ::: "memory");
    }
    __syncthreads();                 // all waves' tile-i data resident in LDS

    Frag16 a0, a1, b0, b1;
    const unsigned int* LA = ldsA[cur];
    const unsigned int* LB = ldsB[cur];
#pragma unroll
    for (int j = 0; j < 8; ++j) {
      int apair = j + ((j >= 4) ? 4 : 0) + 4 * half;
      a0.w[j] = LA[(waveM * 32 + m) * 16 + apair];
      a1.w[j] = LA[(waveM * 32 + 16 + m) * 16 + apair];
      int bpair = j + ((j >= 4) ? 4 : 0) + 4 * half;
      b0.w[j] = LB[(waveN * 32 + m) * 16 + bpair];
      b1.w[j] = LB[(waveN * 32 + 16 + m) * 16 + bpair];
    }
    acc00 = wmma_bf16(a0, b0, acc00);
    acc01 = wmma_bf16(a0, b1, acc01);
    acc10 = wmma_bf16(a1, b0, acc10);
    acc11 = wmma_bf16(a1, b1, acc11);
    __syncthreads();                 // everyone done with buf[cur] before overwrite
  }

  const int c0 = colBase + waveN * 32 + m;
  const int c1 = c0 + 16;
  float bias0 = bias ? bias[c0] : 0.f;
  float bias1 = bias ? bias[c1] : 0.f;
#pragma unroll
  for (int j = 0; j < 8; ++j) {
    int r0 = rowBase + waveM * 32 + j + 8 * half;
    int r1 = r0 + 16;
    C[(size_t)r0 * ldC + c0] = acc00[j] + bias0;
    C[(size_t)r0 * ldC + c1] = acc01[j] + bias1;
    C[(size_t)r1 * ldC + c0] = acc10[j] + bias0;
    C[(size_t)r1 * ldC + c1] = acc11[j] + bias1;
  }
}

// f32 -> packed bf16 conversion, 8 elements/thread (uint4 stores)
__global__ __launch_bounds__(256)
void cvt_bf16_kernel(const float* __restrict__ src, unsigned short* __restrict__ dst,
                     int n8)
{
  int i = blockIdx.x * 256 + threadIdx.x;
  if (i >= n8) return;
  const float4* s = (const float4*)src + (size_t)i * 2;
  float4 v0 = s[0], v1 = s[1];
  uint4 o;
  o.x = pack2bf(v0.x, v0.y);
  o.y = pack2bf(v0.z, v0.w);
  o.z = pack2bf(v1.x, v1.y);
  o.w = pack2bf(v1.z, v1.w);
  ((uint4*)dst)[i] = o;
}

// Small-GEMM K-loop (one wave, one 16x16 tile, convert-on-the-fly) --
// used for the 32-row gathered projections where pre-conversion makes no sense.
__device__ __forceinline__ f32x8 wmma_kloop(const float* __restrict__ aRowLane,
                                            const float* __restrict__ wRowLane,
                                            int Kdim,
                                            unsigned int* ldsA, unsigned int* ldsB,
                                            int lane)
{
  const int rStage = lane >> 1;
  const int seg    = lane & 1;
  const int m      = lane & 15;
  const int half   = lane >> 4;
  f32x8 acc = {};
  for (int kb = 0; kb < Kdim; kb += 32) {
    const float4* as = (const float4*)(aRowLane + kb + seg * 16);
    const float4* ws = (const float4*)(wRowLane + kb + seg * 16);
#pragma unroll
    for (int qq = 0; qq < 4; ++qq) {
      float4 av = as[qq];
      float4 wv = ws[qq];
      int base = rStage * 16 + seg * 8 + 2 * qq;
      ldsA[base]     = pack2bf(av.x, av.y);
      ldsA[base + 1] = pack2bf(av.z, av.w);
      ldsB[base]     = pack2bf(wv.x, wv.y);
      ldsB[base + 1] = pack2bf(wv.z, wv.w);
    }
    __syncthreads();
    Frag16 a, b;
#pragma unroll
    for (int j = 0; j < 8; ++j) {
      a.w[j] = ldsA[m * 16 + j + ((j >= 4) ? 4 : 0) + 4 * half];
      b.w[j] = ldsB[m * 16 + j + ((j >= 4) ? 4 : 0) + 4 * half];
    }
    acc = wmma_bf16(a, b, acc);
    __syncthreads();
  }
  return acc;
}

// Project the 3 candidate selected sets (pos / neg / iota) through Wk and Wv.
// grid.z in [0,6): set = z>>1, K-vs-V = z&1. Six 32x1024x1024 GEMMs total.
__global__ __launch_bounds__(32)
void wmma_proj_sets(const float* __restrict__ MK, const float* __restrict__ MV,
                    const float* __restrict__ WkW, const float* __restrict__ WvW,
                    const int* __restrict__ idxSets,
                    float* __restrict__ Kproj, float* __restrict__ Vproj)
{
  __shared__ unsigned int ldsA[256], ldsB[256];
  const int lane = threadIdx.x;
  const int z = blockIdx.z;
  const int setId = z >> 1, isV = z & 1;
  const int rowBase = blockIdx.y * 16;              // slot tile (0 or 16)
  const int colBase = blockIdx.x * 16;
  const float* A = isV ? MV : MK;
  const float* W = isV ? WvW : WkW;
  float* C = (isV ? Vproj : Kproj) + (size_t)setId * TOPK_K * D_DIM;
  const int aIdx = idxSets[setId * 32 + rowBase + (lane >> 1)];
  const float* aRow = A + (size_t)aIdx * D_DIM;
  const float* wRow = W + (size_t)(colBase + (lane >> 1)) * D_DIM;
  f32x8 acc = wmma_kloop(aRow, wRow, D_DIM, ldsA, ldsB, lane);
  const int n = lane & 15, half = lane >> 4;
#pragma unroll
  for (int j = 0; j < 8; ++j)
    C[(size_t)(rowBase + j + 8 * half) * D_DIM + colBase + n] = acc[j];
}

// r[m] = sum(mk)/||mk||  — monotone surrogate of the cosine score's m-dependence.
// (eps-clamp in the reference only bites when |qm| < ~1e-11: ignored.)
__global__ __launch_bounds__(256)
void mem_stats_kernel(const float* __restrict__ keys, float* __restrict__ r)
{
  __shared__ float s1[256], s2[256];
  const int mrow = blockIdx.x, t = threadIdx.x;
  float4 v = ((const float4*)(keys + (size_t)mrow * D_DIM))[t];
  s1[t] = v.x + v.y + v.z + v.w;
  s2[t] = v.x * v.x + v.y * v.y + v.z * v.z + v.w * v.w;
  __syncthreads();
  for (int s = 128; s > 0; s >>= 1) {
    if (t < s) { s1[t] += s1[t + s]; s2[t] += s2[t + s]; }
    __syncthreads();
  }
  if (t == 0) r[mrow] = s1[0] / fmaxf(sqrtf(s2[0]), 1e-30f);
}

// Single block: top-32 (desc, tie->low idx) and bottom-32 (asc, tie->low idx) of r,
// plus iota fallback for qm==0. 32 rounds of masked arg-reduce; mask = 4KB LDS bitmap.
__global__ __launch_bounds__(256)
void topk_kernel(const float* __restrict__ r, int* __restrict__ idxSets)
{
  __shared__ unsigned int mask[M_MEM / 32];
  __shared__ float sval[256];
  __shared__ int   sidx[256];
  const int t = threadIdx.x;
  if (t < 32) idxSets[64 + t] = t;                  // qm==0 -> top_k of zeros = iota
  for (int phase = 0; phase < 2; ++phase) {
    for (int i = t; i < M_MEM / 32; i += 256) mask[i] = 0u;
    __syncthreads();
    for (int k = 0; k < TOPK_K; ++k) {
      float bv = phase ? __builtin_inff() : -__builtin_inff();
      int   bi = 0x7FFFFFFF;
      for (int mm = t; mm < M_MEM; mm += 256) {
        if (mask[mm >> 5] & (1u << (mm & 31))) continue;
        float v = r[mm];
        bool better = phase ? (v < bv || (v == bv && mm < bi))
                            : (v > bv || (v == bv && mm < bi));
        if (better) { bv = v; bi = mm; }
      }
      sval[t] = bv; sidx[t] = bi;
      __syncthreads();
      for (int s = 128; s > 0; s >>= 1) {
        if (t < s) {
          float ov = sval[t + s]; int oi = sidx[t + s];
          bool better = phase ? (ov < sval[t] || (ov == sval[t] && oi < sidx[t]))
                              : (ov > sval[t] || (ov == sval[t] && oi < sidx[t]));
          if (better) { sval[t] = ov; sidx[t] = oi; }
        }
        __syncthreads();
      }
      if (t == 0) {
        int sel = sidx[0];
        idxSets[phase * 32 + k] = sel;
        mask[sel >> 5] |= (1u << (sel & 31));
      }
      __syncthreads();
    }
    __syncthreads();
  }
}

__global__ __launch_bounds__(256)
void qm_kernel(const float* __restrict__ q_full, float* __restrict__ qm)
{
  __shared__ float s1[256];
  const int p = blockIdx.x, t = threadIdx.x;
  float4 v = ((const float4*)(q_full + (size_t)p * D_DIM))[t];
  s1[t] = v.x + v.y + v.z + v.w;
  __syncthreads();
  for (int s = 128; s > 0; s >>= 1) { if (t < s) s1[t] += s1[t + s]; __syncthreads(); }
  if (t == 0) qm[p] = s1[0] * (1.0f / D_DIM);
}

// Per-position attention over 32 precomputed slots; emits avg_attn, sel_idx,
// and ctx directly in packed bf16 (feeds the async Wo GEMM).
__global__ __launch_bounds__(256)
void attn_kernel(const float* __restrict__ q_full, const float* __restrict__ qm,
                 const float* __restrict__ Kproj, const float* __restrict__ Vproj,
                 const int* __restrict__ idxSets, unsigned short* __restrict__ ctxbf,
                 float* __restrict__ out_avg, float* __restrict__ out_sel)
{
  __shared__ float sc[H_HEADS * TOPK_K];
  const int p = blockIdx.x, t = threadIdx.x;
  const float qv = qm[p];
  const int s = (qv > 0.f) ? 0 : ((qv < 0.f) ? 1 : 2);
  const float* Kp = Kproj + (size_t)s * TOPK_K * D_DIM;
  const float* Vp = Vproj + (size_t)s * TOPK_K * D_DIM;
  const float* q  = q_full + (size_t)p * D_DIM;

  if (t < TOPK_K)                                  // sel_idx written as float values
    out_sel[(size_t)p * TOPK_K + t] = (float)idxSets[s * 32 + t];

#pragma unroll
  for (int rep = 0; rep < 2; ++rep) {              // 512 (h,slot) score dots
    int pi = t + rep * 256;
    int h = pi >> 5, slot = pi & 31;
    const float* qh = q + h * HD;
    const float* kh = Kp + (size_t)slot * D_DIM + h * HD;
    float acc = 0.f;
#pragma unroll 8
    for (int d = 0; d < HD; ++d) acc += qh[d] * kh[d];
    sc[pi] = acc * 0.125f;                         // / (sqrt(64)*TEMPERATURE)
  }
  __syncthreads();

  if (t < H_HEADS) {                               // softmax per head over 32 slots
    float* row = sc + t * TOPK_K;
    float mx = row[0];
    for (int i = 1; i < TOPK_K; ++i) mx = fmaxf(mx, row[i]);
    float sum = 0.f;
    for (int i = 0; i < TOPK_K; ++i) { float e = __expf(row[i] - mx); row[i] = e; sum += e; }
    float inv = 1.f / sum;
    for (int i = 0; i < TOPK_K; ++i) row[i] *= inv;
  }
  __syncthreads();

  if (t < TOPK_K) {
    float a = 0.f;
    for (int h = 0; h < H_HEADS; ++h) a += sc[h * TOPK_K + t];
    out_avg[(size_t)p * TOPK_K + t] = a * (1.0f / H_HEADS);
  }

  {                                                // ctx: 4 contiguous outputs/thread
    int o0 = t * 4;
    int h = o0 >> 6;                               // same head for all 4 (o0 % 64 <= 60)
    const float* vb = Vp + o0;
    float a0 = 0.f, a1 = 0.f, a2 = 0.f, a3 = 0.f;
#pragma unroll 8
    for (int slot = 0; slot < TOPK_K; ++slot) {
      float w = sc[h * TOPK_K + slot];
      const float* vp = vb + (size_t)slot * D_DIM;
      a0 += w * vp[0]; a1 += w * vp[1]; a2 += w * vp[2]; a3 += w * vp[3];
    }
    unsigned int* dst = (unsigned int*)(ctxbf + (size_t)p * D_DIM) + t * 2;
    dst[0] = pack2bf(a0, a1);
    dst[1] = pack2bf(a2, a3);
  }
}

// In-place LayerNorm over d_out rows (biased variance, matching jnp.var).
__global__ __launch_bounds__(256)
void ln_kernel(float* __restrict__ out, const float* __restrict__ gamma,
               const float* __restrict__ beta)
{
  __shared__ float s1[256], s2[256];
  const int p = blockIdx.x, t = threadIdx.x;
  float* row = out + (size_t)p * D_DIM;
  float x[4]; float sum = 0.f, sq = 0.f;
#pragma unroll
  for (int j = 0; j < 4; ++j) { x[j] = row[t + 256 * j]; sum += x[j]; sq += x[j] * x[j]; }
  s1[t] = sum; s2[t] = sq;
  __syncthreads();
  for (int s = 128; s > 0; s >>= 1) {
    if (t < s) { s1[t] += s1[t + s]; s2[t] += s2[t + s]; }
    __syncthreads();
  }
  float mu = s1[0] * (1.f / D_DIM);
  float var = s2[0] * (1.f / D_DIM) - mu * mu;
  float rstd = rsqrtf(var + 1e-5f);
#pragma unroll
  for (int j = 0; j < 4; ++j) {
    int o = t + 256 * j;
    row[o] = (x[j] - mu) * rstd * gamma[o] + beta[o];
  }
}

extern "C" void kernel_launch(void* const* d_in, const int* in_sizes, int n_in,
                              void* d_out, int out_size, void* d_ws, size_t ws_size,
                              hipStream_t stream)
{
  (void)in_sizes; (void)n_in; (void)out_size; (void)ws_size;
  const float* X     = (const float*)d_in[0];   // query_states [4,512,1024]
  const float* MK    = (const float*)d_in[1];   // memory_keys  [32768,1024]
  const float* MV    = (const float*)d_in[2];   // memory_values[32768,1024]
  const float* Wq    = (const float*)d_in[3];
  const float* Wk    = (const float*)d_in[4];
  const float* Wv    = (const float*)d_in[5];
  const float* Wo    = (const float*)d_in[6];
  const float* bo    = (const float*)d_in[7];
  const float* gamma = (const float*)d_in[8];
  const float* beta  = (const float*)d_in[9];
  float* out = (float*)d_out;

  // Workspace layout (~22 MB total)
  char* ws = (char*)d_ws;
  float*          r       = (float*)         (ws + 0x0000000);  // 128 KB
  int*            idxSets = (int*)           (ws + 0x0020000);  // 3 sets x 32
  float*          qm      = (float*)         (ws + 0x0020400);  // 8 KB
  float*          Kproj   = (float*)         (ws + 0x0024000);  // 384 KB
  float*          Vproj   = (float*)         (ws + 0x0084000);  // 384 KB
  float*          q_full  = (float*)         (ws + 0x00E4000);  // 8 MB
  unsigned short* Xbf     = (unsigned short*)(ws + 0x08E4000);  // 4 MB
  unsigned short* Wqbf    = (unsigned short*)(ws + 0x0CE4000);  // 2 MB
  unsigned short* Wobf    = (unsigned short*)(ws + 0x0EE4000);  // 2 MB
  unsigned short* ctxbf   = (unsigned short*)(ws + 0x10E4000);  // 4 MB

  float* out_main = out;                                  // [BN, D]
  float* out_avg  = out + (size_t)BN * D_DIM;             // [BN, 32]
  float* out_sel  = out_avg + (size_t)BN * TOPK_K;        // [BN, 32]

  // 1) per-row stats of memory_keys -> ranking score r[m]   (HBM-bound: 128 MB)
  mem_stats_kernel<<<M_MEM, 256, 0, stream>>>(MK, r);
  // 2) top-32 / bottom-32 / iota index sets (selection depends only on sign(qm))
  topk_kernel<<<1, 256, 0, stream>>>(r, idxSets);
  // 3) one-shot bf16 conversion of GEMM operands
  cvt_bf16_kernel<<<(BN * D_DIM / 8 + 255) / 256, 256, 0, stream>>>(X, Xbf, BN * D_DIM / 8);
  cvt_bf16_kernel<<<(D_DIM * D_DIM / 8 + 255) / 256, 256, 0, stream>>>(Wq, Wqbf, D_DIM * D_DIM / 8);
  cvt_bf16_kernel<<<(D_DIM * D_DIM / 8 + 255) / 256, 256, 0, stream>>>(Wo, Wobf, D_DIM * D_DIM / 8);
  // 4) q_full = X @ Wq^T  (async-staged, double-buffered bf16 WMMA)
  wmma_gemm_bf16_async<<<dim3(D_DIM / BT_N, BN / BT_M), 256, 0, stream>>>(
      Xbf, Wqbf, nullptr, q_full, D_DIM, D_DIM);
  // 5) qm = mean(q_full, -1)
  qm_kernel<<<BN, 256, 0, stream>>>(q_full, qm);
  // 6) K/V projections of the three candidate sets (six 32x1024x1024 WMMA GEMMs)
  wmma_proj_sets<<<dim3(D_DIM / 16, TOPK_K / 16, 6), 32, 0, stream>>>(
      MK, MV, Wk, Wv, idxSets, Kproj, Vproj);
  // 7) per-position attention -> ctx (bf16), avg_attn, sel_idx
  attn_kernel<<<BN, 256, 0, stream>>>(q_full, qm, Kproj, Vproj, idxSets,
                                      ctxbf, out_avg, out_sel);
  // 8) out = ctx @ Wo^T + bo  (async-staged, double-buffered bf16 WMMA)
  wmma_gemm_bf16_async<<<dim3(D_DIM / BT_N, BN / BT_M), 256, 0, stream>>>(
      ctxbf, Wobf, bo, out_main, D_DIM, D_DIM);
  // 9) LayerNorm in place
  ln_kernel<<<BN, 256, 0, stream>>>(out_main, gamma, beta);
}
